// ConnectivityAwareLayer_7335804142017
// MI455X (gfx1250) — matrix operand order repaired and checked by reading, not compiled, not run on hardware
//
#include <hip/hip_runtime.h>

#define D 128
#define NNODES 50000
#define NEDGES 500000

typedef __attribute__((ext_vector_type(8)))  __bf16 v8bf;
typedef __attribute__((ext_vector_type(16))) __bf16 v16bf;
typedef __attribute__((ext_vector_type(8)))  float  v8f;

static __device__ inline v16bf combine8(v8bf a, v8bf b) {
    v16bf r;
#pragma unroll
    for (int i = 0; i < 8; ++i) { r[i] = a[i]; r[i + 8] = b[i]; }
    return r;
}

static __device__ inline v8bf load8(const __bf16* p) {
    return *(const v8bf*)p;
}

static __device__ inline v8bf cvt8(const float* p) {
    v8f f = *(const v8f*)p;
    v8bf r;
#pragma unroll
    for (int i = 0; i < 8; ++i) r[i] = (__bf16)f[i];
    return r;
}

static __device__ inline v8f splat8(float b) {
    v8f c = {b, b, b, b, b, b, b, b};
    return c;
}

#define WMMA_BF16(A, B, C) \
    __builtin_amdgcn_wmma_f32_16x16x32_bf16(false, (A), false, (B), (short)0, (C), false, false)

// ---------------------------------------------------------------- small kernels

__global__ void cvt_bf16_kernel(const float* __restrict__ x, __bf16* __restrict__ o, long n) {
    long i = (long)blockIdx.x * blockDim.x + threadIdx.x;
    if (i < n) o[i] = (__bf16)x[i];
}

__global__ void colsum_kernel(const float* __restrict__ x, float* __restrict__ colsum, int n) {
    int j = threadIdx.x;            // 0..127
    int r0 = blockIdx.x * 256;
    float acc = 0.f;
    int rend = r0 + 256 > n ? n : r0 + 256;
    for (int r = r0; r < rend; ++r) acc += x[(long)r * D + j];
    atomicAdd(&colsum[j], acc);
}

__global__ void global_pool_kernel(const float* __restrict__ colsum,
                                   const float* __restrict__ glbW,
                                   const float* __restrict__ glbb,
                                   float* __restrict__ g, float invN) {
    int j = threadIdx.x;
    float acc = glbb[j];
    for (int k = 0; k < D; ++k) acc += (colsum[k] * invN) * glbW[k * D + j];
    g[j] = fmaxf(acc, 0.f);
}

// transpose + convert weights: W[K][N] f32 -> Wt[N][K] bf16
__global__ void prep_weights_kernel(const float* __restrict__ msgW1, const float* __restrict__ attW1,
                                    const float* __restrict__ updW1,
                                    const float* __restrict__ msgW2, const float* __restrict__ updW2,
                                    __bf16* __restrict__ msgW1t, __bf16* __restrict__ attW1t,
                                    __bf16* __restrict__ updW1t,
                                    __bf16* __restrict__ msgW2t, __bf16* __restrict__ updW2t) {
    int i = blockIdx.x * 256 + threadIdx.x;
    if (i < 2 * D * D) {                  // 256 x 128
        int k = i >> 7, n = i & 127;
        msgW1t[n * 256 + k] = (__bf16)msgW1[i];
        attW1t[n * 256 + k] = (__bf16)attW1[i];
        updW1t[n * 256 + k] = (__bf16)updW1[i];
    }
    if (i < D * D) {                      // 128 x 128
        int k = i >> 7, n = i & 127;
        msgW2t[n * 128 + k] = (__bf16)msgW2[i];
        updW2t[n * 128 + k] = (__bf16)updW2[i];
    }
}

// ---------------------------------------------------------------- edge kernel
// One wave handles 16 edges: A = [h[src] | h[dst]] (16x256 bf16),
// messages = relu(A*W1+b1)*W2+b2, att = sigmoid(mlp), scatter att*msg into agg.

#define LROW 136   // padded LDS row stride (bf16 elements)

__global__ void __launch_bounds__(128)
edge_kernel(const __bf16* __restrict__ hb,
            const long long* __restrict__ src, const long long* __restrict__ dst,
            const __bf16* __restrict__ msgW1t, const __bf16* __restrict__ msgW2t,
            const __bf16* __restrict__ attW1t,
            const float* __restrict__ msg_b1, const float* __restrict__ msg_b2,
            const float* __restrict__ att_b1, const float* __restrict__ attW2,
            const float* __restrict__ att_b2,
            float* __restrict__ agg, int numTiles) {
    __shared__ __bf16 tile[4][16 * LROW];
    __shared__ float attS[4][16];

    const int wave = threadIdx.x >> 5;
    const int lane = threadIdx.x & 31;
    const int t = blockIdx.x * 4 + wave;
    if (t >= numTiles) return;

    const int e0  = t * 16;
    const int row = lane & 15;
    const int hi  = lane >> 4;

    const int sN = (int)src[e0 + row];
    const int dN = (int)dst[e0 + row];
    const __bf16* sp = hb + (long)sN * D;
    const __bf16* dp = hb + (long)dN * D;

    // A fragments for the 16x256 edge-feature matrix (8 K-tiles of 32)
    v16bf A[8];
#pragma unroll
    for (int kt = 0; kt < 4; ++kt) {
        int kb = kt * 32 + hi * 16;
        A[kt] = combine8(load8(sp + kb), load8(sp + kb + 8));
    }
#pragma unroll
    for (int kt = 4; kt < 8; ++kt) {
        int kb = (kt - 4) * 32 + hi * 16;
        A[kt] = combine8(load8(dp + kb), load8(dp + kb + 8));
    }

    __bf16* T = tile[wave];

    // ---- attention layer 1: relu(A * attW1 + b1) -> LDS
#pragma unroll
    for (int nt = 0; nt < 8; ++nt) {
        v8f c = splat8(att_b1[nt * 16 + row]);
#pragma unroll
        for (int kt = 0; kt < 8; ++kt) {
            const __bf16* wp = attW1t + (long)(nt * 16 + row) * 256 + kt * 32 + hi * 16;
            c = WMMA_BF16(A[kt], *(const v16bf*)wp, c);
        }
#pragma unroll
        for (int v = 0; v < 8; ++v) {
            float f = fmaxf(c[v], 0.f);
            T[(v + 8 * hi) * LROW + nt * 16 + row] = (__bf16)f;
        }
    }

    // ---- attention layer 2 (D -> 1) + sigmoid, one lane per edge row
    if (lane < 16) {
        float acc = att_b2[0];
        for (int k = 0; k < D; ++k) acc += (float)T[lane * LROW + k] * attW2[k];
        attS[wave][lane] = 1.f / (1.f + __expf(-acc));
    }

    // ---- message layer 1: relu(A * msgW1 + b1) -> LDS (overwrite)
#pragma unroll
    for (int nt = 0; nt < 8; ++nt) {
        v8f c = splat8(msg_b1[nt * 16 + row]);
#pragma unroll
        for (int kt = 0; kt < 8; ++kt) {
            const __bf16* wp = msgW1t + (long)(nt * 16 + row) * 256 + kt * 32 + hi * 16;
            c = WMMA_BF16(A[kt], *(const v16bf*)wp, c);
        }
#pragma unroll
        for (int v = 0; v < 8; ++v) {
            float f = fmaxf(c[v], 0.f);
            T[(v + 8 * hi) * LROW + nt * 16 + row] = (__bf16)f;
        }
    }

    // ---- message layer 2 + weighted scatter-add
    v16bf A2[4];
#pragma unroll
    for (int kt = 0; kt < 4; ++kt) {
        int kb = kt * 32 + hi * 16;
        A2[kt] = combine8(load8(T + row * LROW + kb), load8(T + row * LROW + kb + 8));
    }

    int   dn[8];
    float av[8];
#pragma unroll
    for (int v = 0; v < 8; ++v) {
        int M = v + 8 * hi;
        dn[v] = (int)dst[e0 + M];
        av[v] = attS[wave][M];
    }

#pragma unroll
    for (int nt = 0; nt < 8; ++nt) {
        v8f c = splat8(msg_b2[nt * 16 + row]);
#pragma unroll
        for (int kt = 0; kt < 4; ++kt) {
            const __bf16* wp = msgW2t + (long)(nt * 16 + row) * 128 + kt * 32 + hi * 16;
            c = WMMA_BF16(A2[kt], *(const v16bf*)wp, c);
        }
#pragma unroll
        for (int v = 0; v < 8; ++v) {
            atomicAdd(&agg[(long)dn[v] * D + nt * 16 + row], c[v] * av[v]);
        }
    }
}

// ---------------------------------------------------------------- node update kernel
// One wave handles 16 nodes: A = [h | agg] (16x256), new_h = mlp2(A) + h + g.

__global__ void __launch_bounds__(128)
update_kernel(const float* __restrict__ hcur, const __bf16* __restrict__ hbcur,
              const float* __restrict__ agg,
              const __bf16* __restrict__ updW1t, const __bf16* __restrict__ updW2t,
              const float* __restrict__ upd_b1, const float* __restrict__ upd_b2,
              const float* __restrict__ g,
              float* __restrict__ hnext, __bf16* __restrict__ hbnext, int numTiles) {
    __shared__ __bf16 tile[4][16 * LROW];

    const int wave = threadIdx.x >> 5;
    const int lane = threadIdx.x & 31;
    const int t = blockIdx.x * 4 + wave;
    if (t >= numTiles) return;

    const int n0  = t * 16;
    const int row = lane & 15;
    const int hi  = lane >> 4;

    const __bf16* hp = hbcur + (long)(n0 + row) * D;
    const float*  ap = agg   + (long)(n0 + row) * D;

    v16bf A[8];
#pragma unroll
    for (int kt = 0; kt < 4; ++kt) {
        int kb = kt * 32 + hi * 16;
        A[kt] = combine8(load8(hp + kb), load8(hp + kb + 8));
    }
#pragma unroll
    for (int kt = 4; kt < 8; ++kt) {
        int kb = (kt - 4) * 32 + hi * 16;
        A[kt] = combine8(cvt8(ap + kb), cvt8(ap + kb + 8));
    }

    __bf16* T = tile[wave];

    // layer 1 -> LDS
#pragma unroll
    for (int nt = 0; nt < 8; ++nt) {
        v8f c = splat8(upd_b1[nt * 16 + row]);
#pragma unroll
        for (int kt = 0; kt < 8; ++kt) {
            const __bf16* wp = updW1t + (long)(nt * 16 + row) * 256 + kt * 32 + hi * 16;
            c = WMMA_BF16(A[kt], *(const v16bf*)wp, c);
        }
#pragma unroll
        for (int v = 0; v < 8; ++v) {
            float f = fmaxf(c[v], 0.f);
            T[(v + 8 * hi) * LROW + nt * 16 + row] = (__bf16)f;
        }
    }

    // layer 2 + residual + global
    v16bf A2[4];
#pragma unroll
    for (int kt = 0; kt < 4; ++kt) {
        int kb = kt * 32 + hi * 16;
        A2[kt] = combine8(load8(T + row * LROW + kb), load8(T + row * LROW + kb + 8));
    }

#pragma unroll
    for (int nt = 0; nt < 8; ++nt) {
        v8f c = splat8(upd_b2[nt * 16 + row]);
#pragma unroll
        for (int kt = 0; kt < 4; ++kt) {
            const __bf16* wp = updW2t + (long)(nt * 16 + row) * 128 + kt * 32 + hi * 16;
            c = WMMA_BF16(A2[kt], *(const v16bf*)wp, c);
        }
        int col = nt * 16 + row;
        float gv = g[col];
#pragma unroll
        for (int v = 0; v < 8; ++v) {
            int node = n0 + v + 8 * hi;
            float res = c[v] + hcur[(long)node * D + col] + gv;
            hnext[(long)node * D + col]  = res;
            hbnext[(long)node * D + col] = (__bf16)res;
        }
    }
}

// ---------------------------------------------------------------- launch

extern "C" void kernel_launch(void* const* d_in, const int* in_sizes, int n_in,
                              void* d_out, int out_size, void* d_ws, size_t ws_size,
                              hipStream_t stream) {
    const float*     x      = (const float*)d_in[0];
    const long long* eidx   = (const long long*)d_in[1];
    const float*     msgW1  = (const float*)d_in[2];
    const float*     msg_b1 = (const float*)d_in[3];
    const float*     msgW2  = (const float*)d_in[4];
    const float*     msg_b2 = (const float*)d_in[5];
    const float*     updW1  = (const float*)d_in[6];
    const float*     upd_b1 = (const float*)d_in[7];
    const float*     updW2  = (const float*)d_in[8];
    const float*     upd_b2 = (const float*)d_in[9];
    const float*     attW1  = (const float*)d_in[10];
    const float*     att_b1 = (const float*)d_in[11];
    const float*     attW2  = (const float*)d_in[12];
    const float*     att_b2 = (const float*)d_in[13];
    const float*     glbW   = (const float*)d_in[14];
    const float*     glbb   = (const float*)d_in[15];

    const long long* src = eidx;
    const long long* dst = eidx + NEDGES;

    char* ws = (char*)d_ws;
    size_t off = 0;
    auto carve = [&](size_t bytes) -> char* {
        char* p = ws + off;
        off = (off + bytes + 255) & ~(size_t)255;
        return p;
    };
    float*  agg    = (float*)carve((size_t)NNODES * D * 4);
    float*  hA     = (float*)carve((size_t)NNODES * D * 4);
    __bf16* hb0    = (__bf16*)carve((size_t)NNODES * D * 2);
    __bf16* hb1    = (__bf16*)carve((size_t)NNODES * D * 2);
    __bf16* msgW1t = (__bf16*)carve((size_t)2 * D * D * 2);
    __bf16* attW1t = (__bf16*)carve((size_t)2 * D * D * 2);
    __bf16* updW1t = (__bf16*)carve((size_t)2 * D * D * 2);
    __bf16* msgW2t = (__bf16*)carve((size_t)D * D * 2);
    __bf16* updW2t = (__bf16*)carve((size_t)D * D * 2);
    float*  colsum = (float*)carve(D * 4);
    float*  gvec   = (float*)carve(D * 4);

    const long nElem = (long)NNODES * D;
    const int edgeTiles = NEDGES / 16;          // 31250
    const int nodeTiles = NNODES / 16 + (NNODES % 16 ? 1 : 0); // 3125
    const int edgeBlocks = (edgeTiles + 3) / 4;
    const int nodeBlocks = (nodeTiles + 3) / 4;

    // prep: colsum zero, hb0 = bf16(x), weight transpose, global repr
    hipMemsetAsync(colsum, 0, D * 4, stream);
    cvt_bf16_kernel<<<(int)((nElem + 255) / 256), 256, 0, stream>>>(x, hb0, nElem);
    colsum_kernel<<<(NNODES + 255) / 256, D, 0, stream>>>(x, colsum, NNODES);
    global_pool_kernel<<<1, D, 0, stream>>>(colsum, glbW, glbb, gvec, 1.0f / (float)NNODES);
    prep_weights_kernel<<<(2 * D * D + 255) / 256, 256, 0, stream>>>(
        msgW1, attW1, updW1, msgW2, updW2, msgW1t, attW1t, updW1t, msgW2t, updW2t);

    // ---- step 0: x -> hA
    hipMemsetAsync(agg, 0, (size_t)NNODES * D * 4, stream);
    edge_kernel<<<edgeBlocks, 128, 0, stream>>>(hb0, src, dst, msgW1t, msgW2t, attW1t,
                                                msg_b1, msg_b2, att_b1, attW2, att_b2,
                                                agg, edgeTiles);
    update_kernel<<<nodeBlocks, 128, 0, stream>>>(x, hb0, agg, updW1t, updW2t,
                                                  upd_b1, upd_b2, gvec, hA, hb1, nodeTiles);

    // ---- step 1: hA -> d_out
    hipMemsetAsync(agg, 0, (size_t)NNODES * D * 4, stream);
    edge_kernel<<<edgeBlocks, 128, 0, stream>>>(hb1, src, dst, msgW1t, msgW2t, attW1t,
                                                msg_b1, msg_b2, att_b1, attW2, att_b2,
                                                agg, edgeTiles);
    update_kernel<<<nodeBlocks, 128, 0, stream>>>(hA, hb1, agg, updW1t, updW2t,
                                                  upd_b1, upd_b2, gvec, (float*)d_out, hb0,
                                                  nodeTiles);
    (void)in_sizes; (void)n_in; (void)out_size; (void)ws_size;
}
